// PrefixSelfAttention_47227460386833
// MI455X (gfx1250) — matrix operand order repaired and checked by reading, not compiled, not run on hardware
//
#include <hip/hip_runtime.h>

typedef __bf16 bf16;
typedef __attribute__((ext_vector_type(16))) __bf16 v16bf;
typedef __attribute__((ext_vector_type(8)))  __bf16 v8bf;
typedef __attribute__((ext_vector_type(8)))  float  v8f;

#define B_ 4
#define T_ 2048
#define C_ 1024
#define H_ 16
#define D_ 64
#define P_ 64
#define S_ (P_ + T_)   // 2112
#define DV_ 80         // D + 16 (row 64 = ones for softmax denominator)

__device__ __forceinline__ v8f wmma_bf16(v16bf a, v16bf b, v8f c) {
  return __builtin_amdgcn_wmma_f32_16x16x32_bf16(false, a, false, b, (short)0, c,
                                                 false, false);
}

// A-fragment (16x32, bf16): lane L holds row L&15, k-chunks {(L>>4)*8 .. +7}
// and {16+(L>>4)*8 .. +7}  (ISA 7.12.2, 16-bit A 16x32 table).
__device__ __forceinline__ v16bf load_frag_a(const bf16* base, int ld) {
  const int lane = threadIdx.x & 31;
  const int row  = lane & 15;
  const int koff = (lane >> 4) << 3;
  const bf16* p = base + row * ld + koff;
  v8bf lo = *(const v8bf*)(p);
  v8bf hi = *(const v8bf*)(p + 16);
  v16bf r;
#pragma unroll
  for (int i = 0; i < 8; ++i) { r[i] = lo[i]; r[i + 8] = hi[i]; }
  return r;
}

// B-fragment (32x16, bf16): lane L holds column L&15 (a row of the row-major
// source), k-chunks {(L>>4)*8 .. +7} and {16+(L>>4)*8 .. +7} (matches A).
__device__ __forceinline__ v16bf load_frag_b(const bf16* base, int ld) {
  const int lane = threadIdx.x & 31;
  const int col  = lane & 15;
  const int koff = (lane >> 4) << 3;
  const bf16* p = base + col * ld + koff;
  v8bf lo = *(const v8bf*)(p);
  v8bf hi = *(const v8bf*)(p + 16);
  v16bf r;
#pragma unroll
  for (int i = 0; i < 8; ++i) { r[i] = lo[i]; r[i + 8] = hi[i]; }
  return r;
}

// ---------------- conversion / init kernels ----------------

__global__ void cvt_kernel(const float* __restrict__ in, bf16* __restrict__ out,
                           int nelem) {
  int i = blockIdx.x * blockDim.x + threadIdx.x;
  if (i < nelem) out[i] = (bf16)in[i];
}

// prefix_k -> Kc[s<P], prefix_v -> Vt (transposed).  i over B*H*P*D.
__global__ void prefix_kernel(const float* __restrict__ pk,
                              const float* __restrict__ pv,
                              bf16* __restrict__ Kc, bf16* __restrict__ Vt) {
  int i = blockIdx.x * blockDim.x + threadIdx.x;
  if (i >= B_ * H_ * P_ * D_) return;
  int d  = i & (D_ - 1);
  int p  = (i >> 6) & (P_ - 1);
  int bh = i >> 12;
  Kc[((size_t)bh * S_ + p) * D_ + d]        = (bf16)pk[i];
  Vt[((size_t)bh * DV_ + d) * S_ + p]       = (bf16)pv[i];
}

// Fill Vt rows d=64..79: row 64 = 1.0 (softmax denominator trick), rest = 0.
__global__ void ones_kernel(bf16* __restrict__ Vt) {
  int i = blockIdx.x * blockDim.x + threadIdx.x;  // over B*H * 16 * S
  const int total = B_ * H_ * 16 * S_;
  if (i >= total) return;
  int s  = i % S_;
  int r  = (i / S_) % 16;          // 0..15 -> d = 64+r
  int bh = i / (16 * S_);
  Vt[((size_t)bh * DV_ + 64 + r) * S_ + s] = (bf16)((r == 0) ? 1.f : 0.f);
}

// ---------------- fused QKV projection GEMM ----------------
// out[m,n] = sum_k x[m,k] * W[n,k] + bias[n]   (x @ W^T + b)
// Each wave: 32M x 64N (2x4 accumulators, 8 wmma / k-step, ~21 FLOP/byte).
// grid (B*T/32, C/256, 3), block 128 (4 waves).

__global__ void __launch_bounds__(128)
qkv_gemm_kernel(const bf16* __restrict__ xb,
                const bf16* __restrict__ Wq, const bf16* __restrict__ Wk,
                const bf16* __restrict__ Wv,
                const float* __restrict__ bq, const float* __restrict__ bk,
                const float* __restrict__ bv,
                bf16* __restrict__ Qb, bf16* __restrict__ Kc,
                bf16* __restrict__ Vt) {
  const int wave  = threadIdx.x >> 5;
  const int lane  = threadIdx.x & 31;
  const int mtile = blockIdx.x;                 // 32-row tile
  const int nt64  = blockIdx.y * 4 + wave;      // 64-col tile == head index
  const int which = blockIdx.z;                 // 0=Q 1=K 2=V

  const bf16*  W    = (which == 0) ? Wq : (which == 1) ? Wk : Wv;
  const float* bias = (which == 0) ? bq : (which == 1) ? bk : bv;

  const bf16* arow0 = xb + (size_t)mtile * 32 * C_;
  const bf16* arow1 = arow0 + 16 * C_;
  const bf16* brow  = W + (size_t)nt64 * 64 * C_;

  v8f acc[2][4] = {};
#pragma unroll 2
  for (int k = 0; k < C_; k += 32) {
    v16bf a0 = load_frag_a(arow0 + k, C_);
    v16bf a1 = load_frag_a(arow1 + k, C_);
#pragma unroll
    for (int ni = 0; ni < 4; ++ni) {
      v16bf b = load_frag_b(brow + (size_t)ni * 16 * C_ + k, C_);
      acc[0][ni] = wmma_bf16(a0, b, acc[0][ni]);
      acc[1][ni] = wmma_bf16(a1, b, acc[1][ni]);
    }
  }

  const int n     = lane & 15;
  const int rbase = (lane >> 4) << 3;
  const int h     = nt64;
#pragma unroll
  for (int mi = 0; mi < 2; ++mi) {
#pragma unroll
    for (int ni = 0; ni < 4; ++ni) {
      const int d     = ni * 16 + n;
      const float bi  = bias[h * 64 + d];
      const int mglob0 = mtile * 32 + mi * 16 + rbase;
      const int bb = mglob0 / T_;
      const int t0 = mglob0 % T_;
#pragma unroll
      for (int j = 0; j < 8; ++j) {
        bf16 v = (bf16)(acc[mi][ni][j] + bi);
        int  t = t0 + j;
        if (which == 0)
          Qb[(((size_t)bb * H_ + h) * T_ + t) * D_ + d] = v;
        else if (which == 1)
          Kc[(((size_t)bb * H_ + h) * S_ + P_ + t) * D_ + d] = v;
        else
          Vt[(((size_t)bb * H_ + h) * DV_ + d) * S_ + P_ + t] = v;
      }
    }
  }
}

// ---------------- flash attention ----------------
// One wave per 16-row query tile; 32-key tiles; online softmax.
// Softmax denominator accumulated by a 5th PV wmma against the ones-row of Vt.
// grid (B*H*T/16/4), block 128.

__global__ void __launch_bounds__(128)
attn_kernel(const bf16* __restrict__ Qb, const bf16* __restrict__ Kc,
            const bf16* __restrict__ Vt, const int* __restrict__ mask,
            bf16* __restrict__ AOb) {
  __shared__ bf16 lds_p[4][16 * 32];
  const int wave = threadIdx.x >> 5;
  const int lane = threadIdx.x & 31;
  const int wgid = blockIdx.x * 4 + wave;
  const int tiles_per_seq = T_ / 16;
  const int bh = wgid / tiles_per_seq;
  const int t0 = (wgid % tiles_per_seq) * 16;
  const int bb = bh / H_;
  const int h  = bh % H_;

  const bf16* qbase = Qb + (((size_t)bh) * T_ + t0) * D_;
  const bf16* kbase = Kc + ((size_t)bh) * S_ * D_;
  const bf16* vbase = Vt + ((size_t)bh) * DV_ * S_;
  const int*  mrow  = mask + bb * T_;

  v16bf qa0 = load_frag_a(qbase, D_);        // d = 0..31
  v16bf qa1 = load_frag_a(qbase + 32, D_);   // d = 32..63

  v8f o0 = {}, o1 = {}, o2 = {}, o3 = {}, o4 = {};   // o4 col 0 = denominator
  float mrun[8];
#pragma unroll
  for (int j = 0; j < 8; ++j) mrun[j] = -__builtin_inff();

  const int n     = lane & 15;
  const int rbase = (lane >> 4) << 3;
  bf16* pbuf = &lds_p[wave][0];

  for (int s0 = 0; s0 < S_; s0 += 32) {
    // key-position mask (prefix always visible)
    const int sg0 = s0 + n, sg1 = s0 + 16 + n;
    const float madd0 = (sg0 < P_ || mrow[sg0 - P_] != 0) ? 0.f : -__builtin_inff();
    const float madd1 = (sg1 < P_ || mrow[sg1 - P_] != 0) ? 0.f : -__builtin_inff();

    // scores = Q . K^T  (K-dim = D = 64 -> 2 wmma per 16x16 score tile)
    v16bf kb00 = load_frag_b(kbase + (size_t)s0 * D_, D_);
    v16bf kb01 = load_frag_b(kbase + (size_t)s0 * D_ + 32, D_);
    v16bf kb10 = load_frag_b(kbase + (size_t)(s0 + 16) * D_, D_);
    v16bf kb11 = load_frag_b(kbase + (size_t)(s0 + 16) * D_ + 32, D_);
    v8f sc0 = {}, sc1 = {};
    sc0 = wmma_bf16(qa0, kb00, sc0);
    sc0 = wmma_bf16(qa1, kb01, sc0);
    sc1 = wmma_bf16(qa0, kb10, sc1);
    sc1 = wmma_bf16(qa1, kb11, sc1);

    // online softmax: only the row-max needs a shuffle tree
    // (rows 0-7 live in lanes 0-15, rows 8-15 in lanes 16-31)
    float p0[8], p1[8], alpha[8];
#pragma unroll
    for (int j = 0; j < 8; ++j) {
      float a = sc0[j] * 0.125f + madd0;   // 1/sqrt(64)
      float c = sc1[j] * 0.125f + madd1;
      float mx = fmaxf(a, c);
#pragma unroll
      for (int off = 1; off < 16; off <<= 1)
        mx = fmaxf(mx, __shfl_xor(mx, off, 32));
      float mnew = fmaxf(mrun[j], mx);
      alpha[j] = __expf(mrun[j] - mnew);
      mrun[j]  = mnew;
      p0[j] = __expf(a - mnew);
      p1[j] = __expf(c - mnew);
    }

    // C-layout -> A-layout transpose of P through wave-private LDS tile
#pragma unroll
    for (int j = 0; j < 8; ++j) {
      int m = rbase + j;
      pbuf[m * 32 + n]      = (bf16)p0[j];
      pbuf[m * 32 + 16 + n] = (bf16)p1[j];
      o0[j] *= alpha[j]; o1[j] *= alpha[j];
      o2[j] *= alpha[j]; o3[j] *= alpha[j];
      o4[j] *= alpha[j];
    }
    v16bf pa = load_frag_a(pbuf, 32);

    // O += P . V   (V transposed: B-fragments contiguous in s)
    o0 = wmma_bf16(pa, load_frag_b(vbase + (size_t)0 * S_  + s0, S_), o0);
    o1 = wmma_bf16(pa, load_frag_b(vbase + (size_t)16 * S_ + s0, S_), o1);
    o2 = wmma_bf16(pa, load_frag_b(vbase + (size_t)32 * S_ + s0, S_), o2);
    o3 = wmma_bf16(pa, load_frag_b(vbase + (size_t)48 * S_ + s0, S_), o3);
    o4 = wmma_bf16(pa, load_frag_b(vbase + (size_t)64 * S_ + s0, S_), o4);
  }

#pragma unroll
  for (int j = 0; j < 8; ++j) {
    // denominator for row rbase+j sits in column n==0 of o4 within this half
    float l   = __shfl(o4[j], lane & 16, 32);
    float inv = 1.f / l;
    int   t   = t0 + rbase + j;
    size_t orow = ((size_t)bb * T_ + t) * C_ + (size_t)h * D_;
    AOb[orow + 0  + n] = (bf16)(o0[j] * inv);
    AOb[orow + 16 + n] = (bf16)(o1[j] * inv);
    AOb[orow + 32 + n] = (bf16)(o2[j] * inv);
    AOb[orow + 48 + n] = (bf16)(o3[j] * inv);
  }
}

// ---------------- output projection ----------------
// Each wave: 32M x 64N, fp32 output with bias.

__global__ void __launch_bounds__(128)
out_gemm_kernel(const bf16* __restrict__ AOb, const bf16* __restrict__ Wo,
                const float* __restrict__ bo, float* __restrict__ out) {
  const int wave  = threadIdx.x >> 5;
  const int lane  = threadIdx.x & 31;
  const int mtile = blockIdx.x;
  const int nt64  = blockIdx.y * 4 + wave;
  const bf16* arow0 = AOb + (size_t)mtile * 32 * C_;
  const bf16* arow1 = arow0 + 16 * C_;
  const bf16* brow  = Wo + (size_t)nt64 * 64 * C_;

  v8f acc[2][4] = {};
#pragma unroll 2
  for (int k = 0; k < C_; k += 32) {
    v16bf a0 = load_frag_a(arow0 + k, C_);
    v16bf a1 = load_frag_a(arow1 + k, C_);
#pragma unroll
    for (int ni = 0; ni < 4; ++ni) {
      v16bf b = load_frag_b(brow + (size_t)ni * 16 * C_ + k, C_);
      acc[0][ni] = wmma_bf16(a0, b, acc[0][ni]);
      acc[1][ni] = wmma_bf16(a1, b, acc[1][ni]);
    }
  }

  const int n = lane & 15, rbase = (lane >> 4) << 3;
#pragma unroll
  for (int mi = 0; mi < 2; ++mi) {
#pragma unroll
    for (int ni = 0; ni < 4; ++ni) {
      const int nglob = nt64 * 64 + ni * 16 + n;
      const float bi  = bo[nglob];
      const size_t m0 = (size_t)mtile * 32 + mi * 16 + rbase;
#pragma unroll
      for (int j = 0; j < 8; ++j)
        out[(m0 + j) * C_ + nglob] = acc[mi][ni][j] + bi;
    }
  }
}

// ---------------- host launch ----------------

extern "C" void kernel_launch(void* const* d_in, const int* in_sizes, int n_in,
                              void* d_out, int out_size, void* d_ws,
                              size_t ws_size, hipStream_t stream) {
  const float* x    = (const float*)d_in[0];
  const int*   amsk = (const int*)d_in[1];
  const float* pk   = (const float*)d_in[2];
  const float* pv   = (const float*)d_in[3];
  const float* Wq   = (const float*)d_in[4];
  const float* bq   = (const float*)d_in[5];
  const float* Wk   = (const float*)d_in[6];
  const float* bk   = (const float*)d_in[7];
  const float* Wv   = (const float*)d_in[8];
  const float* bv   = (const float*)d_in[9];
  const float* Wo   = (const float*)d_in[10];
  const float* bo   = (const float*)d_in[11];
  float* out = (float*)d_out;

  char* ws = (char*)d_ws;
  size_t off = 0;
  auto take = [&](size_t bytes) {
    void* p = ws + off;
    off = (off + bytes + 255) & ~(size_t)255;
    return p;
  };
  const size_t NX = (size_t)B_ * T_ * C_;         // 8388608
  const size_t NW = (size_t)C_ * C_;              // 1048576
  const size_t NK = (size_t)B_ * H_ * S_ * D_;    // 8650752
  const size_t NV = (size_t)B_ * H_ * DV_ * S_;   // 10813440

  bf16* xb  = (bf16*)take(NX * 2);
  bf16* Wqb = (bf16*)take(NW * 2);
  bf16* Wkb = (bf16*)take(NW * 2);
  bf16* Wvb = (bf16*)take(NW * 2);
  bf16* Wob = (bf16*)take(NW * 2);
  bf16* Qb  = (bf16*)take(NX * 2);
  bf16* Kc  = (bf16*)take(NK * 2);
  bf16* Vt  = (bf16*)take(NV * 2);
  bf16* AOb = (bf16*)take(NX * 2);
  (void)ws_size; (void)n_in; (void)in_sizes; (void)out_size;

  cvt_kernel<<<(int)((NX + 255) / 256), 256, 0, stream>>>(x, xb, (int)NX);
  cvt_kernel<<<(int)((NW + 255) / 256), 256, 0, stream>>>(Wq, Wqb, (int)NW);
  cvt_kernel<<<(int)((NW + 255) / 256), 256, 0, stream>>>(Wk, Wkb, (int)NW);
  cvt_kernel<<<(int)((NW + 255) / 256), 256, 0, stream>>>(Wv, Wvb, (int)NW);
  cvt_kernel<<<(int)((NW + 255) / 256), 256, 0, stream>>>(Wo, Wob, (int)NW);

  const int NO = B_ * H_ * 16 * S_;
  ones_kernel<<<(NO + 255) / 256, 256, 0, stream>>>(Vt);

  const int NP = B_ * H_ * P_ * D_;
  prefix_kernel<<<(NP + 255) / 256, 256, 0, stream>>>(pk, pv, Kc, Vt);

  qkv_gemm_kernel<<<dim3(B_ * T_ / 32, C_ / 256, 3), 128, 0, stream>>>(
      xb, Wqb, Wkb, Wvb, bq, bk, bv, Qb, Kc, Vt);

  attn_kernel<<<B_ * H_ * (T_ / 16) / 4, 128, 0, stream>>>(Qb, Kc, Vt, amsk, AOb);

  out_gemm_kernel<<<dim3(B_ * T_ / 32, C_ / 256), 128, 0, stream>>>(AOb, Wob, bo, out);
}
